// VersorRotorRNN_12713103196452
// MI455X (gfx1250) — matrix operand order repaired and checked by reading, not compiled, not run on hardware
//
#include <hip/hip_runtime.h>
#include <hip/hip_bf16.h>

typedef __attribute__((ext_vector_type(16))) _Float16 v16h;
typedef __attribute__((ext_vector_type(8)))  float    v8f;

#define BB 16
#define SS 256
#define NN 16
#define DD 6
#define HH 16

// ---- Cl(5,0) sign table: bit k of SIGNMASK[i] == 1 iff sgn(i, i^k) < 0 ----
constexpr unsigned popc_c(unsigned x) { unsigned c = 0; while (x) { c += x & 1u; x >>= 1; } return c; }
constexpr int sgn_neg(unsigned i, unsigned b) {
  unsigned a = i >> 1, s = 0;
  while (a) { s += popc_c(a & b); a >>= 1; }
  return (int)(s & 1u);
}
constexpr unsigned maskrow(unsigned i) {
  unsigned m = 0;
  for (unsigned k = 0; k < 32; ++k)
    if (sgn_neg(i, i ^ k)) m |= (1u << k);
  return m;
}
__device__ constexpr unsigned SIGNMASK[32] = {
  maskrow(0),  maskrow(1),  maskrow(2),  maskrow(3),
  maskrow(4),  maskrow(5),  maskrow(6),  maskrow(7),
  maskrow(8),  maskrow(9),  maskrow(10), maskrow(11),
  maskrow(12), maskrow(13), maskrow(14), maskrow(15),
  maskrow(16), maskrow(17), maskrow(18), maskrow(19),
  maskrow(20), maskrow(21), maskrow(22), maskrow(23),
  maskrow(24), maskrow(25), maskrow(26), maskrow(27),
  maskrow(28), maskrow(29), maskrow(30), maskrow(31)
};

__device__ __forceinline__ float wave_sum32(float v) {
  v += __shfl_xor(v, 16, 32);
  v += __shfl_xor(v, 8, 32);
  v += __shfl_xor(v, 4, 32);
  v += __shfl_xor(v, 2, 32);
  v += __shfl_xor(v, 1, 32);
  return v;
}

__global__ __launch_bounds__(1024)
void versor_rnn_kernel(const float* __restrict__ x,
                       const float* __restrict__ W_in,
                       const float* __restrict__ b_in,
                       const float* __restrict__ W_out,
                       const float* __restrict__ b_out,
                       float* __restrict__ out) {
  __shared__ float sPSI[2 * HH * 32];    // [chain-local][h][blade]
  __shared__ float sPart[HH * 2 * 16];   // [k-chunk][chain-local][col]

  const int tid  = threadIdx.x;
  const int lane = tid & 31;
  const int w    = tid >> 5;          // wave id 0..31
  const int nl   = w >> 4;            // local chain 0/1
  const int h    = w & 15;            // hidden channel owned by this wave
  const int bp   = blockIdx.x;        // 0..127
  const int b    = bp >> 3;
  const int n0   = (bp & 7) * 2;
  const int n    = n0 + nl;

  // ---- per-lane constant preloads ----
  float win[DD];
#pragma unroll
  for (int d = 0; d < DD; ++d) win[d] = W_in[d * (HH * 32) + h * 32 + lane];
  const float binl = b_in[h * 32 + lane];

  // ---- B fragment for proj_out WMMA (chunk h): B is 32x16 f16
  //      lanes 0-15 : N=lane,    K = {0..7, 16..23}
  //      lanes16-31 : N=lane-16, K = {8..15, 24..31}
  union { v16h v; _Float16 e[16]; } Bf;
#pragma unroll
  for (int j = 0; j < 16; ++j) Bf.e[j] = (_Float16)0.0f;
  {
    const int Ncol  = (lane < 16) ? lane : lane - 16;
    const int kbase = (lane < 16) ? 0 : 8;
    if (Ncol < DD) {
#pragma unroll
      for (int j = 0; j < 16; ++j)
        Bf.e[j] = (_Float16)W_out[(h * 32 + kbase + j + (j < 8 ? 0 : 8)) * DD + Ncol];
    }
  }

  const unsigned shl = 31u - (unsigned)lane;

  // psi0 = identity rotor (scalar blade = 1)
  float psi = (lane == 0) ? 1.0f : 0.0f;

  for (int t = 0; t < SS; ++t) {
    // ---- proj_in (exact f32): u = b_in + x_t . W_in column (6 FMAs) ----
    const float* xp = x + (((b * SS) + t) * NN + n) * DD;
    float u = binl;
#pragma unroll
    for (int d = 0; d < DD; ++d) u = fmaf(xp[d], win[d], u);

    // ---- delta_r = normalize(u + e0) ----
    float dv = u + ((lane == 0) ? 1.0f : 0.0f);
    float s1 = wave_sum32(dv * dv);
    float dr = dv * rsqrtf(s1 + 1e-8f);

    // ---- geometric product via XOR lane shuffles:
    //      acc[k] = sum_i sgn(i, i^k) * dr[i] * psi[i^k]
    float acc = 0.0f;
#pragma unroll
    for (int i = 0; i < 32; ++i) {
      float pj = __shfl_xor(psi, i, 32);      // psi[lane ^ i]
      float di = __shfl(dr, i, 32);           // broadcast dr[i]
      unsigned flip = (SIGNMASK[i] << shl) & 0x80000000u;
      float sd = __uint_as_float(__float_as_uint(di) ^ flip);
      acc = fmaf(sd, pj, acc);
    }
    float s2 = wave_sum32(acc * acc);
    psi = acc * rsqrtf(s2 + 1e-8f);

    // publish psi for the projection GEMM
    sPSI[(nl * HH + h) * 32 + lane] = psi;
    __syncthreads();

    // ---- proj_out via V_WMMA_F32_16X16X32_F16, K-split across waves 0..15
    //      (uniform branch so EXEC is all-ones for executing waves)
    if (__builtin_amdgcn_readfirstlane(w) < 16) {
      // A fragment: 16x32 f16; rows = chains (only 0,1 live), K-chunk = h = w
      //   lanes 0-15 : M=lane,    K = {0..7, 16..23}
      //   lanes16-31 : M=lane-16, K = {8..15, 24..31}
      union { v16h v; _Float16 e[16]; } Af;
#pragma unroll
      for (int j = 0; j < 16; ++j) Af.e[j] = (_Float16)0.0f;
      const int M = (lane < 16) ? lane : lane - 16;
      if (M < 2) {
        const float* pr = &sPSI[(M * HH + w) * 32];
        const int k0 = (lane < 16) ? 0 : 8;
        const int k1 = (lane < 16) ? 16 : 24;
#pragma unroll
        for (int j = 0; j < 8; ++j) {
          Af.e[j]     = (_Float16)pr[k0 + j];
          Af.e[8 + j] = (_Float16)pr[k1 + j];
        }
      }
      v8f c = {0.f, 0.f, 0.f, 0.f, 0.f, 0.f, 0.f, 0.f};
      v8f dacc = __builtin_amdgcn_wmma_f32_16x16x32_f16(
          false, Af.v, false, Bf.v, (short)0, c, false, false);
      // D layout: VGPR r -> row r (lanes 0-15, col=lane); rows 0,1 are our chains
      if (lane < 16) {
        sPart[(w * 2 + 0) * 16 + lane] = dacc[0];
        sPart[(w * 2 + 1) * 16 + lane] = dacc[1];
      }
    }
    __syncthreads();

    // ---- deterministic tree-free reduction + epilogue: y = x + psi@W_out + b_out
    if (tid < 2 * DD) {
      const int nl2 = tid / DD;
      const int d   = tid % DD;
      const int gi  = (((b * SS) + t) * NN + (n0 + nl2)) * DD + d;
      float y = b_out[d] + x[gi];
#pragma unroll
      for (int w2 = 0; w2 < 16; ++w2) y += sPart[(w2 * 2 + nl2) * 16 + d];
      out[gi] = y;
    }
    __syncthreads();
  }
}

extern "C" void kernel_launch(void* const* d_in, const int* in_sizes, int n_in,
                              void* d_out, int out_size, void* d_ws, size_t ws_size,
                              hipStream_t stream) {
  (void)in_sizes; (void)n_in; (void)out_size; (void)d_ws; (void)ws_size;
  const float* x     = (const float*)d_in[0];
  const float* W_in  = (const float*)d_in[1];
  const float* b_in  = (const float*)d_in[2];
  const float* W_out = (const float*)d_in[3];
  const float* b_out = (const float*)d_in[4];
  float* out = (float*)d_out;
  versor_rnn_kernel<<<dim3(128), dim3(1024), 0, stream>>>(x, W_in, b_in, W_out, b_out, out);
}
